// NewModel_29205777613209
// MI455X (gfx1250) — matrix operand order repaired and checked by reading, not compiled, not run on hardware
//
#include <hip/hip_runtime.h>
#include <hip/hip_bf16.h>
#include <math.h>

// ---------------- problem constants ----------------
constexpr int cB = 64, cL = 512, cE = 256, cHID = 256, cH = 128, cG = 512, cT = 5;
constexpr int cBL = cB * cL;          // 32768 rows
constexpr int cMT = cBL / 16;         // 2048 M tiles
constexpr int cNG = cG / 64;          // 8 N-groups of 4 tiles (pre-gemm blocking)

typedef __attribute__((ext_vector_type(16))) _Float16 v16h;
typedef __attribute__((ext_vector_type(8)))  float    v8f;

__device__ inline v8f wmma_f16(v16h a, v16h b, v8f c) {
    // D = A(16x32 f16) * B(32x16 f16) + C(16x16 f32)
    return __builtin_amdgcn_wmma_f32_16x16x32_f16(
        false, a, false, b, (short)0, c, false, false);
}

__device__ inline float sigm(float x) { return 1.0f / (1.0f + expf(-x)); }

// A-matrix fragment: row M = lane%16; lanes<16 take K {kb..kb+7, kb+16..kb+23},
// lanes>=16 take K {kb+8..kb+15, kb+24..kb+31}. Two contiguous 16B loads.
__device__ inline v16h load_a(const _Float16* row, int kb, int lane) {
    v16h a;
    int aoff = (lane < 16) ? 0 : 8;
    uint4* ap = (uint4*)&a;
    ap[0] = *(const uint4*)(row + kb + aoff);
    ap[1] = *(const uint4*)(row + kb + aoff + 16);
    return a;
}
// B-matrix fragment: col N = lane%16; lanes<16 take K {kb..kb+7, kb+16..kb+23},
// lanes>=16 take K {kb+8..kb+15, kb+24..kb+31} (2x16B loads).
__device__ inline v16h load_b(const _Float16* row, int kb, int lane) {
    v16h b;
    int boff = (lane < 16) ? 0 : 8;
    uint4* bp = (uint4*)&b;
    bp[0] = *(const uint4*)(row + kb + boff);
    bp[1] = *(const uint4*)(row + kb + boff + 16);
    return b;
}

// ---------------- kernel 1: doc vector (masked mean of embeddings) ----------
__global__ void k_doc(const int* __restrict__ batch, const int* __restrict__ mask,
                      const float* __restrict__ emb, float* __restrict__ doc) {
    int b = blockIdx.x;
    int e = threadIdx.x;                 // blockDim.x == cE == 256
    __shared__ int cnt_s[256];
    int c = 0;
    for (int l = threadIdx.x; l < cL; l += 256) c += mask[b * cL + l];
    cnt_s[threadIdx.x] = c;
    __syncthreads();
    for (int s = 128; s > 0; s >>= 1) {
        if (threadIdx.x < s) cnt_s[threadIdx.x] += cnt_s[threadIdx.x + s];
        __syncthreads();
    }
    float inv = 1.0f / (float)cnt_s[0];
    float acc = 0.0f;
    for (int l = 0; l < cL; ++l) {
        int id = batch[b * cL + l];
        acc += emb[(size_t)id * cE + e];
    }
    doc[b * cE + e] = acc * inv;
}

// ---------------- kernel 2: x = emb + doc, cast to f16 ----------------------
__global__ void k_buildx(const int* __restrict__ batch, const float* __restrict__ emb,
                         const float* __restrict__ doc, _Float16* __restrict__ x16) {
    size_t n = (size_t)cBL * cE;
    for (size_t i = (size_t)blockIdx.x * blockDim.x + threadIdx.x; i < n;
         i += (size_t)gridDim.x * blockDim.x) {
        int e = (int)(i % cE);
        size_t bl = i / cE;
        int b = (int)(bl / cL);
        int id = batch[bl];
        x16[i] = (_Float16)(emb[(size_t)id * cE + e] + doc[b * cE + e]);
    }
}

// ---------------- kernel 3: weight conversion / packing ---------------------
__global__ void k_cvt(const float* wihf, const float* whhf, const float* bihf, const float* bhhf,
                      const float* wihb, const float* whhb, const float* bihb, const float* bhhb,
                      const float* Wout,
                      _Float16* wih16, _Float16* whh16, _Float16* wpad16, float* biasc) {
    constexpr int N1 = cG * cE;   // 131072
    constexpr int N2 = cG * cH;   // 65536
    constexpr int NP = 16 * cHID; // 4096
    constexpr int NB = 2 * cG;    // 1024
    constexpr int TOT = 2 * N1 + 2 * N2 + NP + NB;
    for (int i = blockIdx.x * blockDim.x + threadIdx.x; i < TOT;
         i += gridDim.x * blockDim.x) {
        int j = i;
        if (j < N1) { wih16[j] = (_Float16)wihf[j]; continue; }
        j -= N1;
        if (j < N1) { wih16[N1 + j] = (_Float16)wihb[j]; continue; }
        j -= N1;
        if (j < N2) { whh16[j] = (_Float16)whhf[j]; continue; }
        j -= N2;
        if (j < N2) { whh16[N2 + j] = (_Float16)whhb[j]; continue; }
        j -= N2;
        if (j < NP) {
            int r = j / cHID, c = j % cHID;
            wpad16[j] = (r < cT) ? (_Float16)Wout[r * cHID + c] : (_Float16)0.0f;
            continue;
        }
        j -= NP;
        if (j < cG) biasc[j] = bihf[j] + bhhf[j];
        else        biasc[j] = bihb[j - cG] + bhhb[j - cG];
    }
}

// ---------------- kernel 4: pre-gate GEMM (both directions) -----------------
// pre[dir][m][g] = x[m,:] . w_ih[dir][g,:] + (b_ih+b_hh)[dir][g], stored f16.
// Register blocking: each wave owns 1 M-tile x 4 N-tiles so one A fragment
// feeds 4 WMMAs per K-chunk (32 WMMAs / wave).
__global__ void k_pregemm(const _Float16* __restrict__ x16,
                          const _Float16* __restrict__ wih16,
                          const float* __restrict__ biasc,
                          _Float16* __restrict__ pre16) {
    int wave = (int)((blockIdx.x * blockDim.x + threadIdx.x) >> 5);
    int lane = threadIdx.x & 31;
    int dir = wave / (cMT * cNG);
    int rem = wave % (cMT * cNG);
    int mt = rem / cNG, ng = rem % cNG;
    int nlane = lane & 15;
    const _Float16* wih = wih16 + (size_t)dir * cG * cE;
    _Float16* pre = pre16 + (size_t)dir * cBL * cG;

    int ncol[4];
    const _Float16* brow[4];
    v8f acc[4];
    for (int j = 0; j < 4; ++j) {
        ncol[j] = (ng * 4 + j) * 16 + nlane;
        brow[j] = wih + (size_t)ncol[j] * cE;
        float bias = biasc[dir * cG + ncol[j]];
        for (int r = 0; r < 8; ++r) acc[j][r] = bias;
    }
    const _Float16* arow = x16 + (size_t)(mt * 16 + nlane) * cE;
    for (int kc = 0; kc < cE / 32; ++kc) {
        int kb = kc * 32;
        v16h a = load_a(arow, kb, lane);
        for (int j = 0; j < 4; ++j) {
            v16h b = load_b(brow[j], kb, lane);
            acc[j] = wmma_f16(a, b, acc[j]);
        }
    }
    int mbase = mt * 16 + ((lane < 16) ? 0 : 8);
    for (int j = 0; j < 4; ++j)
        for (int r = 0; r < 8; ++r)
            pre[(size_t)(mbase + r) * cG + ncol[j]] = (_Float16)acc[j][r];
}

// ---------------- kernel 5: LSTM recurrence ---------------------------------
// grid = 8 blocks: block = (dir, batch-tile of 16). 8 waves/block, wave = one
// 16-wide hidden tile -> 4 gate tiles (i,f,g,o). h kept in LDS (double buffer),
// c in registers (v8f matches WMMA C layout). 16 WMMAs per wave per step;
// next step's pre tiles are prefetched (global_prefetch_b8) during the GEMM.
__global__ void k_lstm(const _Float16* __restrict__ pre16,
                       const _Float16* __restrict__ whh16,
                       _Float16* __restrict__ h_all) {
    int dir = blockIdx.x >> 2;
    int mt  = blockIdx.x & 3;           // batch tile (16 rows)
    int wv  = threadIdx.x >> 5;         // hidden tile 0..7
    int lane = threadIdx.x & 31;
    int nlane = lane & 15;
    __shared__ _Float16 hbuf[2][16][cH];        // 8 KB double-buffered h
    const _Float16* pre = pre16 + (size_t)dir * cBL * cG;
    const _Float16* whh = whh16 + (size_t)dir * cG * cH;

    for (int i = threadIdx.x; i < 16 * cH; i += blockDim.x)
        hbuf[0][i / cH][i % cH] = (_Float16)0.0f;
    __syncthreads();

    const _Float16* brow[4];
    int ncol[4];
    for (int g = 0; g < 4; ++g) {
        ncol[g] = (g * 8 + wv) * 16 + nlane;
        brow[g] = whh + (size_t)ncol[g] * cH;
    }
    v8f c;
    for (int r = 0; r < 8; ++r) c[r] = 0.0f;
    int mloc0 = (lane < 16) ? 0 : 8;
    int buf = 0;
    for (int s = 0; s < cL; ++s) {
        int t = dir ? (cL - 1 - s) : s;
        v8f acc[4];
        for (int g = 0; g < 4; ++g)
            for (int r = 0; r < 8; ++r) {
                int bglob = mt * 16 + mloc0 + r;
                acc[g][r] = (float)pre[((size_t)bglob * cL + t) * cG + ncol[g]];
            }
        // prefetch next step's pre-activations into cache while we do WMMAs
        if (s + 1 < cL) {
            int tn = dir ? (cL - 2 - s) : (s + 1);
            for (int g = 0; g < 4; ++g)
                __builtin_prefetch(
                    &pre[((size_t)(mt * 16 + mloc0) * cL + tn) * cG + ncol[g]], 0, 0);
        }
        const _Float16* hrow = &hbuf[buf][nlane][0];
        for (int kc = 0; kc < cH / 32; ++kc) {
            int kb = kc * 32;
            v16h a = load_a(hrow, kb, lane);   // ds_load_b128 from LDS
            for (int g = 0; g < 4; ++g) {
                v16h b = load_b(brow[g], kb, lane);
                acc[g] = wmma_f16(a, b, acc[g]);
            }
        }
        int nbuf = buf ^ 1;
        int j = wv * 16 + nlane;
        for (int r = 0; r < 8; ++r) {
            float iv = sigm(acc[0][r]);
            float fv = sigm(acc[1][r]);
            float gv = tanhf(acc[2][r]);
            float ov = sigm(acc[3][r]);
            float cn = fv * c[r] + iv * gv;
            c[r] = cn;
            float hv = ov * tanhf(cn);
            int m = mloc0 + r;
            hbuf[nbuf][m][j] = (_Float16)hv;
            h_all[(((size_t)(mt * 16 + m)) * cL + t) * cHID + dir * cH + j] =
                (_Float16)hv;
        }
        __syncthreads();
        buf = nbuf;
    }
}

// ---------------- kernel 6: output projection (T=5 padded to 16) ------------
__global__ void k_outgemm(const _Float16* __restrict__ h_all,
                          const _Float16* __restrict__ wpad,
                          const float* __restrict__ b_out,
                          float* __restrict__ logits) {
    int wave = (int)((blockIdx.x * blockDim.x + threadIdx.x) >> 5);  // M tile
    int lane = threadIdx.x & 31;
    int nlane = lane & 15;
    float bias = (nlane < cT) ? b_out[nlane] : 0.0f;
    v8f acc;
    for (int r = 0; r < 8; ++r) acc[r] = bias;
    const _Float16* arow = h_all + (size_t)(wave * 16 + nlane) * cHID;
    const _Float16* brow = wpad + (size_t)nlane * cHID;
    for (int kc = 0; kc < cHID / 32; ++kc) {
        int kb = kc * 32;
        v16h a = load_a(arow, kb, lane);
        v16h b = load_b(brow, kb, lane);
        acc = wmma_f16(a, b, acc);
    }
    if (nlane < cT) {
        int mbase = wave * 16 + ((lane < 16) ? 0 : 8);
        for (int r = 0; r < 8; ++r)
            logits[(size_t)(mbase + r) * cT + nlane] = acc[r];
    }
}

// ---------------- kernel 7: CRF NLL (tiny, 1 thread per sequence) -----------
__global__ void k_crf(const float* __restrict__ logits, const int* __restrict__ y,
                      const int* __restrict__ mask, const float* __restrict__ start_t,
                      const float* __restrict__ end_t, const float* __restrict__ trans,
                      float* __restrict__ out) {
    __shared__ float red[cB];
    int b = threadIdx.x;
    const float* lg = logits + (size_t)b * cL * cT;
    const int* tg = y + b * cL;
    const int* mk = mask + b * cL;
    float alpha[cT];
    int t0 = tg[0];
    float score = start_t[t0] + lg[t0];
    for (int j = 0; j < cT; ++j) alpha[j] = start_t[j] + lg[j];
    int cnt = mk[0];
    for (int t = 1; t < cL; ++t) {
        int m = mk[t];
        cnt += m;
        float mf = (float)m;
        int tc = tg[t], tp = tg[t - 1];
        score += (trans[tp * cT + tc] + lg[t * cT + tc]) * mf;
        float nxt[cT];
        for (int j = 0; j < cT; ++j) {
            float mx = -1e30f;
            for (int i = 0; i < cT; ++i)
                mx = fmaxf(mx, alpha[i] + trans[i * cT + j]);
            float s = 0.0f;
            for (int i = 0; i < cT; ++i)
                s += expf(alpha[i] + trans[i * cT + j] - mx);
            nxt[j] = mx + logf(s) + lg[t * cT + j];
        }
        for (int j = 0; j < cT; ++j) alpha[j] = (m > 0) ? nxt[j] : alpha[j];
    }
    score += end_t[tg[cnt - 1]];
    float mx = -1e30f;
    for (int j = 0; j < cT; ++j) mx = fmaxf(mx, alpha[j] + end_t[j]);
    float s = 0.0f;
    for (int j = 0; j < cT; ++j) s += expf(alpha[j] + end_t[j] - mx);
    float logz = mx + logf(s);
    red[b] = score - logz;
    __syncthreads();
    if (b == 0) {
        float sum = 0.0f;
        for (int i = 0; i < cB; ++i) sum += red[i];
        out[0] = -sum / (float)cB;
    }
}

// ---------------- launch ----------------------------------------------------
extern "C" void kernel_launch(void* const* d_in, const int* in_sizes, int n_in,
                              void* d_out, int out_size, void* d_ws, size_t ws_size,
                              hipStream_t stream) {
    const int*   batch = (const int*)d_in[0];
    const int*   y     = (const int*)d_in[1];
    const int*   mask  = (const int*)d_in[2];
    const float* emb   = (const float*)d_in[3];
    const float* wihf  = (const float*)d_in[4];
    const float* whhf  = (const float*)d_in[5];
    const float* bihf  = (const float*)d_in[6];
    const float* bhhf  = (const float*)d_in[7];
    const float* wihb  = (const float*)d_in[8];
    const float* whhb  = (const float*)d_in[9];
    const float* bihb  = (const float*)d_in[10];
    const float* bhhb  = (const float*)d_in[11];
    const float* Wout  = (const float*)d_in[12];
    const float* bout  = (const float*)d_in[13];
    const float* start_t = (const float*)d_in[14];
    const float* end_t   = (const float*)d_in[15];
    const float* trans   = (const float*)d_in[16];
    float* out = (float*)d_out;

    // workspace carve-up (256B aligned)
    char* ws = (char*)d_ws;
    size_t off = 0;
    auto carve = [&](size_t bytes) {
        size_t o = off;
        off = (off + bytes + 255) & ~(size_t)255;
        return o;
    };
    _Float16* x16    = (_Float16*)(ws + carve((size_t)cBL * cE * 2));        // 16 MB
    _Float16* pre16  = (_Float16*)(ws + carve((size_t)2 * cBL * cG * 2));    // 64 MB
    _Float16* h_all  = (_Float16*)(ws + carve((size_t)cBL * cHID * 2));      // 16 MB
    _Float16* wih16  = (_Float16*)(ws + carve((size_t)2 * cG * cE * 2));
    _Float16* whh16  = (_Float16*)(ws + carve((size_t)2 * cG * cH * 2));
    _Float16* wpad16 = (_Float16*)(ws + carve((size_t)16 * cHID * 2));
    float*    biasc  = (float*)(ws + carve((size_t)2 * cG * 4));
    float*    doc    = (float*)(ws + carve((size_t)cB * cE * 4));
    float*    logits = (float*)(ws + carve((size_t)cBL * cT * 4));

    k_doc<<<cB, 256, 0, stream>>>(batch, mask, emb, doc);
    k_buildx<<<8192, 256, 0, stream>>>(batch, emb, doc, x16);
    k_cvt<<<1556, 256, 0, stream>>>(wihf, whhf, bihf, bhhf,
                                    wihb, whhb, bihb, bhhb, Wout,
                                    wih16, whh16, wpad16, biasc);
    // 2 dirs * 2048 Mtiles * 8 Ngroups = 32768 waves / 8 waves per block
    k_pregemm<<<4096, 256, 0, stream>>>(x16, wih16, biasc, pre16);
    k_lstm<<<8, 256, 0, stream>>>(pre16, whh16, h_all);
    // 2048 M tiles / 8 waves per block
    k_outgemm<<<256, 256, 0, stream>>>(h_all, wpad16, bout, logits);
    k_crf<<<1, cB, 0, stream>>>(logits, y, mask, start_t, end_t, trans, out);
}
